// GroupedQueryAttention_1778116460719
// MI455X (gfx1250) — compile-verified, matrix operand order checked by reading, not run on hardware
//
#include <hip/hip_runtime.h>
#include <hip/hip_bf16.h>

// ---------------------------------------------------------------------------
// CDNA5 (gfx1250) GQA attention: WMMA f16 GEMMs + flash attention,
// with TDM (tensor_load_to_lds) and async global->LDS staging where available.
// wave32, 8 waves / 256-thread block.
// ---------------------------------------------------------------------------

typedef __attribute__((ext_vector_type(16))) _Float16     v16h;
typedef __attribute__((ext_vector_type(8)))  float        v8f;
typedef __attribute__((ext_vector_type(4)))  unsigned int u32x4;
typedef __attribute__((ext_vector_type(8)))  int          i32x8;
typedef __attribute__((ext_vector_type(4)))  int          i32x4;
typedef __attribute__((ext_vector_type(4)))  int          v4i;

#if defined(__HIP_DEVICE_COMPILE__) && __has_builtin(__builtin_amdgcn_tensor_load_to_lds)
#define HAVE_TDM 1
#else
#define HAVE_TDM 0
#endif
#if defined(__HIP_DEVICE_COMPILE__) && __has_builtin(__builtin_amdgcn_global_load_async_to_lds_b128)
#define HAVE_ASYNC 1
#else
#define HAVE_ASYNC 0
#endif

#define LDS_VP(p)   ((__attribute__((address_space(3))) void*)(p))
#define GLB_B128(p) ((__attribute__((address_space(1))) v4i*)(void*)(p))
#define LDS_B128(p) ((__attribute__((address_space(3))) v4i*)(p))

union Frag {            // one 16x32 f16 WMMA operand (A or B), 8 VGPRs / lane
    v16h h;
    unsigned int u[8];
};

// K-index pattern for the 16-bit 16x32 A/B fragment layout (ISA 7.12.2):
// lanes 0-15 hold K pairs {0..7, 16..23}; lanes 16-31 hold {8..15, 24..31}.
__device__ __forceinline__ int kpat(int v, int half) {
    return (v < 4) ? (half * 8 + 2 * v) : (16 + half * 8 + 2 * (v - 4));
}

__device__ __forceinline__ v8f wmma_f16(const v16h& a, const v16h& b, const v8f& c) {
    return __builtin_amdgcn_wmma_f32_16x16x32_f16(
        /*neg_a=*/false, a, /*neg_b=*/false, b,
        /*c_mod=*/(short)0, c, /*reuse_a=*/false, /*reuse_b=*/false);
}

// ---------------------------------------------------------------------------
// GEMM: C[M,N] = A[M,K] * B[K,N].  A: float or _Float16, B: float, C: OT.
// Block tile 128x128, K-step 32. 8 waves, each wave owns 16 rows x 128 cols
// (8 chained v_wmma_f32_16x16x32_f16 per K-step).
// ---------------------------------------------------------------------------
#define BM 128
#define BN 128
#define BK 32

template <typename AT, typename OT>
__global__ __launch_bounds__(256)
void gemm_wmma_kernel(const AT* __restrict__ A, const float* __restrict__ B,
                      OT* __restrict__ C, int M, int N, int K)
{
    __shared__ _Float16 Ah[BM][BK];     // 8 KB
    __shared__ _Float16 Bh[BN][BK];     // 8 KB, transposed: [n][k]

    const int tid  = threadIdx.x;
    const int wave = tid >> 5;
    const int lane = tid & 31;
    const int half = lane >> 4;
    const int l16  = lane & 15;
    const int m0   = blockIdx.y * BM;
    const int n0   = blockIdx.x * BN;

    v8f acc[8];
#pragma unroll
    for (int j = 0; j < 8; ++j)
#pragma unroll
        for (int i = 0; i < 8; ++i) acc[j][i] = 0.0f;

    for (int k0 = 0; k0 < K; k0 += BK) {
        // ---- stage A tile (128 x 32) ----
        if constexpr (sizeof(AT) == 2) {
#if HAVE_ASYNC
            // pure f16 copy: async global -> LDS (b128), tracked by ASYNCcnt
#pragma unroll
            for (int i = 0; i < 2; ++i) {
                int slot = tid + i * 256;             // 512 slots of 8 halves
                int row  = slot >> 2;                 // 4 chunks per 64B row
                int c8   = (slot & 3) * 8;
                __builtin_amdgcn_global_load_async_to_lds_b128(
                    GLB_B128(A + (size_t)(m0 + row) * K + k0 + c8),
                    LDS_B128(&Ah[row][c8]), 0, 0);
            }
            asm volatile("s_wait_asynccnt 0" ::: "memory");
#else
#pragma unroll
            for (int i = 0; i < 2; ++i) {
                int slot = tid + i * 256;
                int row  = slot >> 2;
                int c8   = (slot & 3) * 8;
                float4 v = *(const float4*)(A + (size_t)(m0 + row) * K + k0 + c8);
                *(float4*)&Ah[row][c8] = v;
            }
#endif
        } else {
            // fp32 -> f16 conversion while staging
#pragma unroll
            for (int i = 0; i < 4; ++i) {
                int slot = tid + i * 256;             // 1024 slots of 4 floats
                int row  = slot >> 3;                 // BK/4 = 8 slots per row
                int col  = (slot & 7) * 4;
                float4 v = *(const float4*)(A + (size_t)(m0 + row) * K + k0 + col);
                Ah[row][col + 0] = (_Float16)v.x;
                Ah[row][col + 1] = (_Float16)v.y;
                Ah[row][col + 2] = (_Float16)v.z;
                Ah[row][col + 3] = (_Float16)v.w;
            }
        }
        // ---- stage B tile (32 x 128) transposed, fp32 -> f16 ----
#pragma unroll
        for (int i = 0; i < 4; ++i) {
            int slot = tid + i * 256;                 // 1024 slots of 4 floats
            int kk   = slot >> 5;                     // BN/4 = 32 slots per row
            int col  = (slot & 31) * 4;
            float4 v = *(const float4*)(B + (size_t)(k0 + kk) * N + n0 + col);
            Bh[col + 0][kk] = (_Float16)v.x;
            Bh[col + 1][kk] = (_Float16)v.y;
            Bh[col + 2][kk] = (_Float16)v.z;
            Bh[col + 3][kk] = (_Float16)v.w;
        }
        if (k0 + BK < K) {                            // global_prefetch_b8
            __builtin_prefetch(A + (size_t)(m0 + (tid & 127)) * K + k0 + BK, 0, 1);
            __builtin_prefetch(B + (size_t)(k0 + BK + (tid & 31)) * N + n0, 0, 1);
        }
        __syncthreads();

        // ---- compute: wave w -> rows [w*16, w*16+16), cols [0,128) ----
        Frag a;
#pragma unroll
        for (int v = 0; v < 8; ++v) {
            int kk = kpat(v, half);
            a.u[v] = *(const unsigned int*)&Ah[wave * 16 + l16][kk];
        }
#pragma unroll
        for (int j = 0; j < 8; ++j) {
            Frag b;
#pragma unroll
            for (int v = 0; v < 8; ++v) {
                int kk = kpat(v, half);
                b.u[v] = *(const unsigned int*)&Bh[j * 16 + l16][kk];
            }
            acc[j] = wmma_f16(a.h, b.h, acc[j]);
        }
        __syncthreads();
    }

    // ---- epilogue: C layout lane0-15 -> M=r, lane16-31 -> M=r+8 ----
#pragma unroll
    for (int j = 0; j < 8; ++j)
#pragma unroll
        for (int r = 0; r < 8; ++r) {
            int row = m0 + wave * 16 + r + 8 * half;
            int col = n0 + j * 16 + l16;
            C[(size_t)row * N + col] = (OT)acc[j][r];
        }
}

// ---------------------------------------------------------------------------
// RoPE (rotate-half, Llama-style) applied in place on f16 [S, nheads*128].
// ---------------------------------------------------------------------------
__global__ void rope_kernel(_Float16* __restrict__ X, int S, int nheads)
{
    int idx   = blockIdx.x * blockDim.x + threadIdx.x;
    int total = S * nheads * 64;
    if (idx >= total) return;
    int i = idx % 64;
    int h = (idx / 64) % nheads;
    int s = idx / (64 * nheads);

    float inv = __expf(-((float)(2 * i) / 128.0f) * __logf(10000.0f));
    float ang = (float)s * inv;
    float c  = __cosf(ang);
    float sn = __sinf(ang);

    _Float16* p = X + (size_t)s * nheads * 128 + h * 128;
    float x1 = (float)p[i];
    float x2 = (float)p[i + 64];
    p[i]      = (_Float16)(x1 * c - x2 * sn);
    p[i + 64] = (_Float16)(x2 * c + x1 * sn);
}

// ---------------------------------------------------------------------------
// Flash attention, causal, GQA (n_rep = 4).
// Grid: (S/128 query blocks, H heads). 8 waves; each wave owns 16 query rows.
// K tile staged via TDM (tensor_load_to_lds) when available.
// ---------------------------------------------------------------------------
__global__ __launch_bounds__(256)
void fa_kernel(const _Float16* __restrict__ Q, const _Float16* __restrict__ K,
               const _Float16* __restrict__ V, _Float16* __restrict__ O, int S)
{
    __shared__ _Float16 Kt[32][128];        // 8 KB   [key][hd]
    __shared__ _Float16 Vt[128][34];        // 8.5 KB [hd][key] (transposed)
    __shared__ _Float16 Pt[8][16][32];      // 8 KB   per-wave P tiles

    const int h    = blockIdx.y;
    const int kvh  = h >> 2;                // H/HKV = 4
    const int qb   = blockIdx.x;
    const int tid  = threadIdx.x;
    const int wave = tid >> 5;
    const int lane = tid & 31;
    const int half = lane >> 4;
    const int l16  = lane & 15;
    const size_t qstride = 32 * 128;        // H*HD
    const size_t kstride = 8 * 128;         // HKV*HD
    const int qrow = qb * 128 + wave * 16 + l16;
    const float scale = 0.08838834764831845f;   // 1/sqrt(128)

    // Q fragments for this wave's 16 rows: 4 chunks of HD (K=32 each)
    Frag qa[4];
#pragma unroll
    for (int c = 0; c < 4; ++c)
#pragma unroll
        for (int v = 0; v < 8; ++v) {
            int kk = kpat(v, half);
            qa[c].u[v] = *(const unsigned int*)
                         &Q[(size_t)qrow * qstride + h * 128 + c * 32 + kk];
        }

    v8f o[8];
    float mrun[8], lrun[8];
#pragma unroll
    for (int t = 0; t < 8; ++t) {
#pragma unroll
        for (int i = 0; i < 8; ++i) o[t][i] = 0.0f;
    }
#pragma unroll
    for (int r = 0; r < 8; ++r) { mrun[r] = -3.0e38f; lrun[r] = 0.0f; }

    const int ntiles = qb * 4 + 4;          // 32-key tiles up to the diagonal
    for (int kt = 0; kt < ntiles; ++kt) {
        const int kbase = kt * 32;
        __syncthreads();

        // ---- stage K tile [32 keys x 128 hd] ----
#if HAVE_TDM
        if (wave == 0) {
            // 2D D# descriptor: tile 128x32 f16, row stride = HKV*HD elements.
            unsigned long long ga =
                (unsigned long long)(const void*)(K + (size_t)kbase * kstride + kvh * 128);
            unsigned int la =
                (unsigned int)(unsigned long long)LDS_VP(&Kt[0][0]);
            u32x4 g0;
            g0[0] = 1u;                                   // count=1, user mode
            g0[1] = la;                                   // lds_addr
            g0[2] = (unsigned int)(ga & 0xFFFFFFFFu);     // global_addr[31:0]
            g0[3] = (unsigned int)((ga >> 32) & 0x1FFFFFFu) | (2u << 30); // type=2
            i32x8 g1;
            g1[0] = (1 << 16);        // data_size = 2 bytes, mask=0
            g1[1] = (128 << 16);      // tensor_dim0 = 128 (lo16 at bits 63:48)
            g1[2] = (32 << 16);       // tensor_dim1 = 32  (lo16 at bits 95:80)
            g1[3] = (128 << 16);      // tile_dim0 = 128   (bits 127:112)
            g1[4] = 32;               // tile_dim1 = 32    (bits 143:128)
            g1[5] = 1024;             // tensor_dim0_stride = HKV*HD
            g1[6] = 0;
            g1[7] = 0;
            i32x4 gz = {0, 0, 0, 0};
#if __clang_major__ >= 23
            i32x8 gz8 = {0, 0, 0, 0, 0, 0, 0, 0};
            __builtin_amdgcn_tensor_load_to_lds(g0, g1, gz, gz, gz8, 0);
#else
            __builtin_amdgcn_tensor_load_to_lds(g0, g1, gz, gz, 0);
#endif
        }
#elif HAVE_ASYNC
#pragma unroll
        for (int i = 0; i < 2; ++i) {
            int slot = tid + i * 256;
            int key  = slot >> 4;
            int hd8  = (slot & 15) * 8;
            __builtin_amdgcn_global_load_async_to_lds_b128(
                GLB_B128(K + (size_t)(kbase + key) * kstride + kvh * 128 + hd8),
                LDS_B128(&Kt[key][hd8]), 0, 0);
        }
#else
#pragma unroll
        for (int i = 0; i < 2; ++i) {
            int slot = tid + i * 256;
            int key  = slot >> 4;
            int hd8  = (slot & 15) * 8;
            float4 vv = *(const float4*)
                        (K + (size_t)(kbase + key) * kstride + kvh * 128 + hd8);
            *(float4*)&Kt[key][hd8] = vv;
        }
#endif
        // ---- stage V tile transposed [128 hd x 32 keys] ----
#pragma unroll
        for (int i = 0; i < 2; ++i) {
            int slot = tid + i * 256;
            int key  = slot >> 4;
            int hd8  = (slot & 15) * 8;
            const _Float16* src =
                V + (size_t)(kbase + key) * kstride + kvh * 128 + hd8;
            float2 raw  = *(const float2*)src;
            float2 raw2 = *(const float2*)(src + 4);
            const _Float16* sh  = (const _Float16*)&raw;
            const _Float16* sh2 = (const _Float16*)&raw2;
#pragma unroll
            for (int j = 0; j < 4; ++j) Vt[hd8 + j][key]     = sh[j];
#pragma unroll
            for (int j = 0; j < 4; ++j) Vt[hd8 + 4 + j][key] = sh2[j];
        }
        if (kt + 1 < ntiles)
            __builtin_prefetch(K + (size_t)(kbase + 32 + (tid & 31)) * kstride
                                 + kvh * 128, 0, 1);
#if HAVE_TDM
        __builtin_amdgcn_s_wait_tensorcnt(0);
#elif HAVE_ASYNC
        asm volatile("s_wait_asynccnt 0" ::: "memory");
#endif
        __syncthreads();

        // ---- scores S[16q x 32k] = Q * K^T : 2 C-tiles, 4 chained WMMA each ----
        v8f s0, s1;
#pragma unroll
        for (int i = 0; i < 8; ++i) { s0[i] = 0.0f; s1[i] = 0.0f; }
#pragma unroll
        for (int c = 0; c < 4; ++c) {
            Frag b0, b1;
#pragma unroll
            for (int v = 0; v < 8; ++v) {
                int kk = kpat(v, half);
                b0.u[v] = *(const unsigned int*)&Kt[l16][c * 32 + kk];
                b1.u[v] = *(const unsigned int*)&Kt[16 + l16][c * 32 + kk];
            }
            s0 = wmma_f16(qa[c].h, b0.h, s0);
            s1 = wmma_f16(qa[c].h, b1.h, s1);
        }

        // ---- causal mask + online softmax (row = r + 8*half, col = lane) ----
        float p0[8], p1[8];
#pragma unroll
        for (int r = 0; r < 8; ++r) {
            int qg   = qb * 128 + wave * 16 + r + 8 * half;
            int k0g  = kbase + l16;
            int k1g  = kbase + 16 + l16;
            bool in0 = (k0g <= qg);
            bool in1 = (k1g <= qg);
            float v0 = in0 ? s0[r] * scale : -3.0e38f;
            float v1 = in1 ? s1[r] * scale : -3.0e38f;
            float rowmax = fmaxf(v0, v1);
#pragma unroll
            for (int m = 8; m >= 1; m >>= 1)
                rowmax = fmaxf(rowmax, __shfl_xor(rowmax, m, 32));
            float mnew = fmaxf(mrun[r], rowmax);
            float fac  = __expf(mrun[r] - mnew);
            float e0   = in0 ? __expf(v0 - mnew) : 0.0f;
            float e1   = in1 ? __expf(v1 - mnew) : 0.0f;
            float ls   = e0 + e1;
#pragma unroll
            for (int m = 8; m >= 1; m >>= 1)
                ls += __shfl_xor(ls, m, 32);
            lrun[r] = lrun[r] * fac + ls;
            mrun[r] = mnew;
#pragma unroll
            for (int t = 0; t < 8; ++t) o[t][r] *= fac;
            p0[r] = e0; p1[r] = e1;
        }

        // ---- C-layout -> A-fragment re-layout via per-wave LDS tile ----
#pragma unroll
        for (int r = 0; r < 8; ++r) {
            Pt[wave][r + 8 * half][l16]      = (_Float16)p0[r];
            Pt[wave][r + 8 * half][16 + l16] = (_Float16)p1[r];
        }
        Frag pa;                               // LDS is in-order within a wave
#pragma unroll
        for (int v = 0; v < 8; ++v) {
            int kk = kpat(v, half);
            pa.u[v] = *(const unsigned int*)&Pt[wave][l16][kk];
        }

        // ---- O[16 x 128] += P[16 x 32] @ V[32 x 128] : 8 WMMAs ----
#pragma unroll
        for (int t = 0; t < 8; ++t) {
            Frag b;
#pragma unroll
            for (int v = 0; v < 8; ++v) {
                int kk = kpat(v, half);
                b.u[v] = *(const unsigned int*)&Vt[t * 16 + l16][kk];
            }
            o[t] = wmma_f16(pa.h, b.h, o[t]);
        }
    }

    // ---- normalize and write merged output [S, H*128] f16 ----
#pragma unroll
    for (int t = 0; t < 8; ++t)
#pragma unroll
        for (int r = 0; r < 8; ++r) {
            int row = qb * 128 + wave * 16 + r + 8 * half;
            float val = o[t][r] / lrun[r];
            O[(size_t)row * qstride + h * 128 + t * 16 + l16] = (_Float16)val;
        }
}

// ---------------------------------------------------------------------------
// Host-side orchestration.
// ---------------------------------------------------------------------------
extern "C" void kernel_launch(void* const* d_in, const int* in_sizes, int n_in,
                              void* d_out, int out_size, void* d_ws, size_t ws_size,
                              hipStream_t stream)
{
    (void)in_sizes; (void)n_in; (void)out_size; (void)ws_size;

    const float* x  = (const float*)d_in[0];
    const float* Wq = (const float*)d_in[1];
    const float* Wk = (const float*)d_in[2];
    const float* Wv = (const float*)d_in[3];
    const float* Wo = (const float*)d_in[4];
    float* out = (float*)d_out;

    const int S = 2048, D = 4096, H = 32, HKV = 8, HD = 128;

    char* ws = (char*)d_ws;
    _Float16* Qh = (_Float16*)ws;                               // S*D       f16
    _Float16* Kh = (_Float16*)(ws + (size_t)S * D * 2);         // S*HKV*HD  f16
    _Float16* Vh = (_Float16*)(ws + (size_t)S * D * 2 + (size_t)S * HKV * HD * 2);
    _Float16* Mh = (_Float16*)(ws + (size_t)S * D * 2 + 2 * (size_t)S * HKV * HD * 2);

    dim3 blk(256);

    // 1. QKV projections (fp32 in, f16 out), WMMA GEMMs
    gemm_wmma_kernel<float, _Float16>
        <<<dim3(D / BN, S / BM), blk, 0, stream>>>(x, Wq, Qh, S, D, D);
    gemm_wmma_kernel<float, _Float16>
        <<<dim3((HKV * HD) / BN, S / BM), blk, 0, stream>>>(x, Wk, Kh, S, HKV * HD, D);
    gemm_wmma_kernel<float, _Float16>
        <<<dim3((HKV * HD) / BN, S / BM), blk, 0, stream>>>(x, Wv, Vh, S, HKV * HD, D);

    // 2. RoPE on Q and K
    rope_kernel<<<(S * H * 64 + 255) / 256, 256, 0, stream>>>(Qh, S, H);
    rope_kernel<<<(S * HKV * 64 + 255) / 256, 256, 0, stream>>>(Kh, S, HKV);

    // 3. Causal GQA flash attention -> merged [S, D] f16
    fa_kernel<<<dim3(S / 128, H), blk, 0, stream>>>(Qh, Kh, Vh, Mh, S);

    // 4. Output projection (f16 in, fp32 out)
    gemm_wmma_kernel<_Float16, float>
        <<<dim3(D / BN, S / BM), blk, 0, stream>>>(Mh, Wo, out, S, D, D);
}
